// uniGRU_72550587564388
// MI455X (gfx1250) — compile-verified, matrix operand order checked by reading, not run on hardware
//
#include <hip/hip_runtime.h>
#include <hip/hip_bf16.h>

// ---------------------------------------------------------------------------
// uniGRU on MI455X (gfx1250): bf16 WMMA recurrent cell.
//
// Dims: VOCAB=50000 EMB=300 HID=1024 OUT=20 BATCH=256 SEQ=128
// K split: c-part K=1024, x-part padded 300->320, total K=1344 = 42*32.
//
// Pipeline (all on `stream`, graph-capture safe):
//   pack_wru, pack_wc  : fp32 weights -> bf16, pre-swizzled into WMMA B-frag
//                        layout (32 lanes x 16 bf16 per 32x16 K x N fragment)
//   gather_xe          : embed all tokens once -> bf16 [B, T, 320] (zero pad)
//   init_c             : c0 = 0 (fp32 + bf16 mirrors)
//   per step t (128x):
//     gates_kernel     : r,u = sigmoid([c_prev,x] @ Wru^T + b)   (WMMA bf16)
//                        writes Acbuf = bf16(r * c_prev), Ubuf = u
//     cand_kernel      : cct = tanh([r*c,x] @ Wc^T + b_c)        (WMMA bf16)
//                        c = u*cct + (1-u)*c_prev  -> Cbuf(f32) + cbf(bf16)
//     y_kernel         : y_t = c @ Wp^T + b_p  -> d_out[b][t][0..19]
// ---------------------------------------------------------------------------

typedef __bf16 bf16_t;
typedef unsigned short u16;
typedef __attribute__((ext_vector_type(16))) __bf16 v16bf;
typedef __attribute__((ext_vector_type(8)))  float  v8f;
typedef __attribute__((ext_vector_type(4)))  unsigned int u32x4;

#define HID   1024
#define EMB_P 320          // padded embedding K
#define KTOT  1344         // HID + EMB_P
#define KT    42           // KTOT / 32
#define BATCH 256
#define SEQ   128
#define OUT   20

union FragU { v16bf v; u32x4 u4[2]; };

__device__ inline v8f zero_v8f() {
  v8f z;
#pragma unroll
  for (int i = 0; i < 8; ++i) z[i] = 0.0f;
  return z;
}

__device__ inline unsigned short f2bf(float x) {
  unsigned u = __builtin_bit_cast(unsigned, x);
  unsigned r = u + 0x7FFFu + ((u >> 16) & 1u);   // round-to-nearest-even
  return (unsigned short)(r >> 16);
}

// A fragment (16 M x 32 K, bf16), row-major source with row stride in elems.
// ISA layout: lanes 0-15 hold M=lane, K = k0+[0..7],k0+[16..23];
//             lanes 16-31 hold M=lane-16, K = k0+[8..15],k0+[24..31].
__device__ inline v16bf load_a_frag(const u16* base, int row0, int rowStride,
                                    int k0, int l15, int hi) {
  const u16* p = base + (size_t)(row0 + l15) * rowStride + k0 + hi * 8;
  FragU f;
  f.u4[0] = *(const u32x4*)(p);
  f.u4[1] = *(const u32x4*)(p + 16);
  return f.v;
}

// B fragment (32 K x 16 N, bf16) from pre-packed buffer: 512 elems / fragment,
// 16 contiguous bf16 per lane (one 32B coalesced load per lane).
__device__ inline v16bf load_b_frag(const u16* pack, size_t fragIdx, int lane) {
  const u16* p = pack + fragIdx * 512 + (size_t)lane * 16;
  FragU f;
  f.u4[0] = *(const u32x4*)(p);
  f.u4[1] = *(const u32x4*)(p + 8);
  return f.v;
}

__device__ inline v8f wmma_bf16(v16bf a, v16bf b, v8f c) {
  return __builtin_amdgcn_wmma_f32_16x16x32_bf16(false, a, false, b,
                                                 (short)0, c, false, false);
}

// ------------------------- weight packing ---------------------------------
// Fragment element (lane,e) -> (n,k): n = nTile*16 + (lane&15),
// k = kt*32 + (e&7) + ((e>>3)*16) + (lane>>4)*8  (mirrors A-frag K layout).
__global__ __launch_bounds__(256)
void pack_wru_kernel(u16* __restrict__ pack, const float* __restrict__ Wr,
                     const float* __restrict__ Wu) {
  long idx = (long)blockIdx.x * blockDim.x + threadIdx.x;
  if (idx >= (long)2048 * KTOT) return;
  long fragIdx = idx >> 9;
  int  rem  = (int)(idx & 511);
  int  lane = rem >> 4, e = rem & 15;
  int  nTile = (int)(fragIdx / KT), kt = (int)(fragIdx % KT);
  int  n = nTile * 16 + (lane & 15);
  int  k = kt * 32 + (e & 7) + ((e >> 3) * 16) + ((lane >> 4) * 8);
  float val = 0.0f;
  if (k < 1324) {               // cat = HID + 300; cols [0,1024)=c, [1024,1324)=x
    val = (n < HID) ? Wr[(size_t)n * 1324 + k]
                    : Wu[(size_t)(n - HID) * 1324 + k];
  }
  pack[idx] = f2bf(val);
}

__global__ __launch_bounds__(256)
void pack_wc_kernel(u16* __restrict__ pack, const float* __restrict__ Wc) {
  long idx = (long)blockIdx.x * blockDim.x + threadIdx.x;
  if (idx >= (long)HID * KTOT) return;
  long fragIdx = idx >> 9;
  int  rem  = (int)(idx & 511);
  int  lane = rem >> 4, e = rem & 15;
  int  nTile = (int)(fragIdx / KT), kt = (int)(fragIdx % KT);
  int  n = nTile * 16 + (lane & 15);
  int  k = kt * 32 + (e & 7) + ((e >> 3) * 16) + ((lane >> 4) * 8);
  float val = (k < 1324) ? Wc[(size_t)n * 1324 + k] : 0.0f;
  pack[idx] = f2bf(val);
}

// ------------------------- embedding gather -------------------------------
__global__ __launch_bounds__(256)
void gather_xe_kernel(u16* __restrict__ xe, const int* __restrict__ x,
                      const float* __restrict__ emb) {
  long idx = (long)blockIdx.x * blockDim.x + threadIdx.x;
  if (idx >= (long)BATCH * SEQ * EMB_P) return;
  int e  = (int)(idx % EMB_P);
  long bt = idx / EMB_P;                 // b*SEQ + t
  unsigned short v = 0;
  if (e < 300) {
    int tok = x[bt];
    v = f2bf(emb[(size_t)tok * 300 + e]);
  }
  xe[idx] = v;
}

__global__ __launch_bounds__(256)
void init_c_kernel(float* __restrict__ Cbuf, u16* __restrict__ cbf) {
  long idx = (long)blockIdx.x * blockDim.x + threadIdx.x;
  if (idx >= (long)BATCH * HID) return;
  Cbuf[idx] = 0.0f;
  cbf[idx]  = 0;
}

// ------------------------- gates (r,u) WMMA kernel ------------------------
// grid (4,16), block 256 = 8 waves (2M x 4N), wave tile 32M x 16N, dual gate.
__global__ __launch_bounds__(256)
void gates_kernel(const u16* __restrict__ cbf, const u16* __restrict__ xe,
                  const u16* __restrict__ wru, const float* __restrict__ b_r,
                  const float* __restrict__ b_u, const float* __restrict__ Cbuf,
                  u16* __restrict__ Acbuf, float* __restrict__ Ubuf, int t) {
  const int lane = threadIdx.x & 31;
  const int wave = threadIdx.x >> 5;
  const int l15  = lane & 15;
  const int hi   = lane >> 4;
  const int mBase = blockIdx.x * 64 + (wave & 1) * 32;
  const int nBase = blockIdx.y * 64 + (wave >> 1) * 16;
  const int nTileR = nBase >> 4;           // 0..63
  const int nTileU = nTileR + 64;          // u stacked after r in the pack
  const u16* xebase = xe + (size_t)t * EMB_P;

  v8f accR0 = zero_v8f(), accR1 = zero_v8f();
  v8f accU0 = zero_v8f(), accU1 = zero_v8f();

  // K segment 1: c_prev (K = 0..1023)
  for (int kt = 0; kt < 32; ++kt) {
    const int k0 = kt * 32;
    v16bf a0 = load_a_frag(cbf, mBase,      HID, k0, l15, hi);
    v16bf a1 = load_a_frag(cbf, mBase + 16, HID, k0, l15, hi);
    v16bf br = load_b_frag(wru, (size_t)nTileR * KT + kt, lane);
    v16bf bu = load_b_frag(wru, (size_t)nTileU * KT + kt, lane);
    accR0 = wmma_bf16(a0, br, accR0);
    accR1 = wmma_bf16(a1, br, accR1);
    accU0 = wmma_bf16(a0, bu, accU0);
    accU1 = wmma_bf16(a1, bu, accU1);
  }
  // K segment 2: x_t (K = 1024..1343), row stride SEQ*EMB_P in xe[B,T,320]
  for (int kt = 32; kt < KT; ++kt) {
    const int k0 = (kt - 32) * 32;
    v16bf a0 = load_a_frag(xebase, mBase,      SEQ * EMB_P, k0, l15, hi);
    v16bf a1 = load_a_frag(xebase, mBase + 16, SEQ * EMB_P, k0, l15, hi);
    v16bf br = load_b_frag(wru, (size_t)nTileR * KT + kt, lane);
    v16bf bu = load_b_frag(wru, (size_t)nTileU * KT + kt, lane);
    accR0 = wmma_bf16(a0, br, accR0);
    accR1 = wmma_bf16(a1, br, accR1);
    accU0 = wmma_bf16(a0, bu, accU0);
    accU1 = wmma_bf16(a1, bu, accU1);
  }

  // Epilogue. D layout: lanes 0-15 -> (M=v, N=lane); 16-31 -> (M=v+8, N=l15).
  const int n = nBase + l15;
  const float brv = b_r[n], buv = b_u[n];
#pragma unroll
  for (int v = 0; v < 8; ++v) {
    {
      int b = mBase + v + hi * 8;
      size_t idx = (size_t)b * HID + n;
      float r = 1.0f / (1.0f + __expf(-(accR0[v] + brv)));
      float u = 1.0f / (1.0f + __expf(-(accU0[v] + buv)));
      Acbuf[idx] = f2bf(r * Cbuf[idx]);
      Ubuf[idx]  = u;
    }
    {
      int b = mBase + 16 + v + hi * 8;
      size_t idx = (size_t)b * HID + n;
      float r = 1.0f / (1.0f + __expf(-(accR1[v] + brv)));
      float u = 1.0f / (1.0f + __expf(-(accU1[v] + buv)));
      Acbuf[idx] = f2bf(r * Cbuf[idx]);
      Ubuf[idx]  = u;
    }
  }
}

// ------------------ candidate + state-update WMMA kernel ------------------
__global__ __launch_bounds__(256)
void cand_kernel(const u16* __restrict__ Acbuf, const u16* __restrict__ xe,
                 const u16* __restrict__ wc, const float* __restrict__ b_c,
                 const float* __restrict__ Ubuf, float* __restrict__ Cbuf,
                 u16* __restrict__ cbf, int t) {
  const int lane = threadIdx.x & 31;
  const int wave = threadIdx.x >> 5;
  const int l15  = lane & 15;
  const int hi   = lane >> 4;
  const int mBase = blockIdx.x * 64 + (wave & 1) * 32;
  const int nBase = blockIdx.y * 64 + (wave >> 1) * 16;
  const int nTile = nBase >> 4;
  const u16* xebase = xe + (size_t)t * EMB_P;

  v8f acc0 = zero_v8f(), acc1 = zero_v8f();

  for (int kt = 0; kt < 32; ++kt) {
    const int k0 = kt * 32;
    v16bf a0 = load_a_frag(Acbuf, mBase,      HID, k0, l15, hi);
    v16bf a1 = load_a_frag(Acbuf, mBase + 16, HID, k0, l15, hi);
    v16bf bc = load_b_frag(wc, (size_t)nTile * KT + kt, lane);
    acc0 = wmma_bf16(a0, bc, acc0);
    acc1 = wmma_bf16(a1, bc, acc1);
  }
  for (int kt = 32; kt < KT; ++kt) {
    const int k0 = (kt - 32) * 32;
    v16bf a0 = load_a_frag(xebase, mBase,      SEQ * EMB_P, k0, l15, hi);
    v16bf a1 = load_a_frag(xebase, mBase + 16, SEQ * EMB_P, k0, l15, hi);
    v16bf bc = load_b_frag(wc, (size_t)nTile * KT + kt, lane);
    acc0 = wmma_bf16(a0, bc, acc0);
    acc1 = wmma_bf16(a1, bc, acc1);
  }

  const int n = nBase + l15;
  const float bcv = b_c[n];
#pragma unroll
  for (int v = 0; v < 8; ++v) {
    {
      int b = mBase + v + hi * 8;
      size_t idx = (size_t)b * HID + n;
      float cct = tanhf(acc0[v] + bcv);
      float u = Ubuf[idx], cp = Cbuf[idx];
      float cn = u * cct + (1.0f - u) * cp;
      Cbuf[idx] = cn;
      cbf[idx]  = f2bf(cn);
    }
    {
      int b = mBase + 16 + v + hi * 8;
      size_t idx = (size_t)b * HID + n;
      float cct = tanhf(acc1[v] + bcv);
      float u = Ubuf[idx], cp = Cbuf[idx];
      float cn = u * cct + (1.0f - u) * cp;
      Cbuf[idx] = cn;
      cbf[idx]  = f2bf(cn);
    }
  }
}

// ------------------------- output projection ------------------------------
__global__ __launch_bounds__(256)
void y_kernel(const float* __restrict__ Cbuf, const float* __restrict__ Wp,
              const float* __restrict__ bp, float* __restrict__ out, int t) {
  const int b = blockIdx.x;
  const int tid = threadIdx.x;
  float acc[OUT];
#pragma unroll
  for (int o = 0; o < OUT; ++o) acc[o] = 0.0f;
  for (int h = tid; h < HID; h += 256) {
    float c = Cbuf[(size_t)b * HID + h];
#pragma unroll
    for (int o = 0; o < OUT; ++o) acc[o] += c * Wp[(size_t)o * HID + h];
  }
  __shared__ float red[256];
  for (int o = 0; o < OUT; ++o) {
    red[tid] = acc[o];
    __syncthreads();
    for (int s = 128; s > 0; s >>= 1) {
      if (tid < s) red[tid] += red[tid + s];
      __syncthreads();
    }
    if (tid == 0) out[((size_t)b * SEQ + t) * OUT + o] = red[0] + bp[o];
    __syncthreads();
  }
}

// ---------------------------------------------------------------------------
extern "C" void kernel_launch(void* const* d_in, const int* in_sizes, int n_in,
                              void* d_out, int out_size, void* d_ws, size_t ws_size,
                              hipStream_t stream) {
  const int*   x    = (const int*)  d_in[0];
  const float* emb  = (const float*)d_in[1];
  const float* W_r  = (const float*)d_in[2];
  const float* b_r  = (const float*)d_in[3];
  const float* W_u  = (const float*)d_in[4];
  const float* b_u  = (const float*)d_in[5];
  const float* W_c  = (const float*)d_in[6];
  const float* b_c  = (const float*)d_in[7];
  const float* W_p  = (const float*)d_in[8];
  const float* b_p  = (const float*)d_in[9];
  float* out = (float*)d_out;

  char* ws = (char*)d_ws;
  u16*   wru   = (u16*)  (ws);                       // 2048*1344*2  = 5505024 B
  u16*   wc    = (u16*)  (ws + 5505024);             // 1024*1344*2  = 2752512 B
  u16*   xe    = (u16*)  (ws + 8257536);             // 256*128*320*2= 20971520 B
  float* Cbuf  = (float*)(ws + 29229056);            // 256*1024*4   = 1048576 B
  u16*   cbf   = (u16*)  (ws + 30277632);            // 256*1024*2   = 524288 B
  u16*   Acbuf = (u16*)  (ws + 30801920);            // 256*1024*2   = 524288 B
  float* Ubuf  = (float*)(ws + 31326208);            // 256*1024*4   = 1048576 B
  (void)in_sizes; (void)n_in; (void)out_size; (void)ws_size;

  // One-time per call: pack weights, gather embeddings, zero state.
  {
    long n = (long)2048 * KTOT;
    pack_wru_kernel<<<dim3((unsigned)((n + 255) / 256)), dim3(256), 0, stream>>>(wru, W_r, W_u);
  }
  {
    long n = (long)HID * KTOT;
    pack_wc_kernel<<<dim3((unsigned)((n + 255) / 256)), dim3(256), 0, stream>>>(wc, W_c);
  }
  {
    long n = (long)BATCH * SEQ * EMB_P;
    gather_xe_kernel<<<dim3((unsigned)((n + 255) / 256)), dim3(256), 0, stream>>>(xe, x, emb);
  }
  {
    long n = (long)BATCH * HID;
    init_c_kernel<<<dim3((unsigned)((n + 255) / 256)), dim3(256), 0, stream>>>(Cbuf, cbf);
  }

  // Recurrent sweep: 128 steps x (gates -> candidate/update -> projection).
  const dim3 gemmGrid(4, 16);   // M tiles (256/64) x N tiles (1024/64)
  const dim3 gemmBlock(256);    // 8 waves
  for (int t = 0; t < SEQ; ++t) {
    gates_kernel<<<gemmGrid, gemmBlock, 0, stream>>>(cbf, xe, wru, b_r, b_u,
                                                     Cbuf, Acbuf, Ubuf, t);
    cand_kernel<<<gemmGrid, gemmBlock, 0, stream>>>(Acbuf, xe, wc, b_c,
                                                    Ubuf, Cbuf, cbf, t);
    y_kernel<<<dim3(BATCH), dim3(256), 0, stream>>>(Cbuf, W_p, b_p, out, t);
  }
}
